// MatMulFreeModel_22857815949785
// MI455X (gfx1250) — 8-bit matrix operand order repaired and checked by reading, compiled, not run on hardware
//
#include <hip/hip_runtime.h>
#include <cstdint>

#define DIM   1024
#define BN    4
#define TT    4096
#define LN    4
#define BT    (BN*TT)
#define CH    32
#define CL    (TT/CH)
#define AROW  1040

typedef __attribute__((ext_vector_type(8))) int v8i;

__device__ __forceinline__ float sigm(float x)  { return __builtin_amdgcn_rcpf(1.0f + __expf(-x)); }
__device__ __forceinline__ float siluf(float x) { return x * sigm(x); }

__device__ __forceinline__ float blk_sum256(float v, float* s8) {
  #pragma unroll
  for (int o = 16; o; o >>= 1) v += __shfl_xor(v, o, 32);
  const int wave = threadIdx.x >> 5, lane = threadIdx.x & 31;
  if (lane == 0) s8[wave] = v;
  __syncthreads();
  float t = 0.f;
  #pragma unroll
  for (int w = 0; w < 8; ++w) t += s8[w];
  return t;
}
__device__ __forceinline__ float blk_max256(float v, float* s8) {
  #pragma unroll
  for (int o = 16; o; o >>= 1) v = fmaxf(v, __shfl_xor(v, o, 32));
  const int wave = threadIdx.x >> 5, lane = threadIdx.x & 31;
  if (lane == 0) s8[wave] = v;
  __syncthreads();
  float t = 0.f;
  #pragma unroll
  for (int w = 0; w < 8; ++w) t = fmaxf(t, s8[w]);
  return t;
}

__global__ __launch_bounds__(256) void embed_kernel(const int* __restrict__ tok,
    const float* __restrict__ emb, float* __restrict__ x) {
  const int tid = blockIdx.x * 256 + threadIdx.x;
  const int row = tid >> 8;
  const int c   = tid & 255;
  const int t   = tok[row];
  ((float4*)x)[(size_t)row * 256 + c] =
      ((const float4*)(emb + (size_t)t * DIM))[c];
}

__global__ __launch_bounds__(256) void wq_part(const float* __restrict__ W,
                                               float* __restrict__ part) {
  __shared__ float s8[8];
  float acc = 0.f;
  for (int idx = blockIdx.x * 256 + threadIdx.x; idx < DIM * DIM / 4; idx += 256 * 256) {
    float4 v = ((const float4*)W)[idx];
    acc += fabsf(v.x) + fabsf(v.y) + fabsf(v.z) + fabsf(v.w);
  }
  float t = blk_sum256(acc, s8);
  if (threadIdx.x == 0) part[blockIdx.x] = t;
}
__global__ __launch_bounds__(256) void wq_final(const float* __restrict__ part,
                                                float* __restrict__ sum) {
  __shared__ float s8[8];
  float t = blk_sum256(part[threadIdx.x], s8);
  if (threadIdx.x == 0) *sum = t;
}
__global__ __launch_bounds__(256) void wq_quant(const float* __restrict__ W,
    const float* __restrict__ sum, int8_t* __restrict__ qw) {
  const float meanc = fmaxf(*sum * (1.0f / (float)(DIM * DIM)), 1e-5f);
  const float inv = 1.0f / meanc;
  const int idx = blockIdx.x * 256 + threadIdx.x;
  float4 v = ((const float4*)W)[idx];
  int b0 = (int)fminf(fmaxf(rintf(v.x * inv), -1.f), 1.f);
  int b1 = (int)fminf(fmaxf(rintf(v.y * inv), -1.f), 1.f);
  int b2 = (int)fminf(fmaxf(rintf(v.z * inv), -1.f), 1.f);
  int b3 = (int)fminf(fmaxf(rintf(v.w * inv), -1.f), 1.f);
  unsigned p = (unsigned)(b0 & 255) | ((unsigned)(b1 & 255) << 8) |
               ((unsigned)(b2 & 255) << 16) | ((unsigned)(b3 & 255) << 24);
  ((unsigned*)qw)[idx] = p;
}

__global__ __launch_bounds__(256) void rq_kernel(const float* __restrict__ x,
    const float* __restrict__ nw, int8_t* __restrict__ q, float* __restrict__ ainv) {
  __shared__ float s8[8];
  __shared__ float m8[8];
  const int row = blockIdx.x;
  const int t = threadIdx.x;
  const float4 v = ((const float4*)(x + (size_t)row * DIM))[t];
  float ss = v.x * v.x + v.y * v.y + v.z * v.z + v.w * v.w;
  const float tot = blk_sum256(ss, s8);
  const float r = rsqrtf(tot * (1.0f / (float)DIM) + 1e-6f);
  const float4 nv = ((const float4*)nw)[t];
  const float hx = v.x * r * nv.x, hy = v.y * r * nv.y;
  const float hz = v.z * r * nv.z, hw = v.w * r * nv.w;
  float am = fmaxf(fmaxf(fabsf(hx), fabsf(hy)), fmaxf(fabsf(hz), fabsf(hw)));
  const float gm = blk_max256(am, m8);
  const float sc = 127.0f / fmaxf(gm, 1e-5f);
  int b0 = (int)fminf(fmaxf(rintf(hx * sc), -128.f), 127.f);
  int b1 = (int)fminf(fmaxf(rintf(hy * sc), -128.f), 127.f);
  int b2 = (int)fminf(fmaxf(rintf(hz * sc), -128.f), 127.f);
  int b3 = (int)fminf(fmaxf(rintf(hw * sc), -128.f), 127.f);
  unsigned p = (unsigned)(b0 & 255) | ((unsigned)(b1 & 255) << 8) |
               ((unsigned)(b2 & 255) << 16) | ((unsigned)(b3 & 255) << 24);
  ((unsigned*)q)[(size_t)row * (DIM / 4) + t] = p;
  if (t == 0) ainv[row] = 1.0f / sc;
}

__global__ __launch_bounds__(256) void gemm_kernel(
    const int8_t* __restrict__ qa, const float* __restrict__ ainv,
    const int8_t* __restrict__ qw, const float* __restrict__ wsum,
    float* __restrict__ out, const float* __restrict__ fprev, int mode) {
  __shared__ int8_t smA[16 * AROW];
  const int wave = threadIdx.x >> 5;
  const int lane = threadIdx.x & 31;
  const int hi   = lane >> 4;
  const int l16  = lane & 15;
  const int mtile = blockIdx.x;

  #pragma unroll
  for (int it = 0; it < 4; ++it) {
    const int c = it * 256 + threadIdx.x;
    const int r = c >> 6;
    const int colb = (c & 63) << 4;
    const int8_t* g = qa + (size_t)(mtile * 16 + r) * DIM + colb;
    const unsigned lds = (unsigned)(uintptr_t)(smA + r * AROW + colb);
    asm volatile("global_load_async_to_lds_b128 %0, %1, off"
                 :: "v"(lds), "v"(g) : "memory");
  }
  asm volatile("s_wait_asynccnt 0" ::: "memory");
  __syncthreads();

  const float wsc = fmaxf(*wsum * (1.0f / (float)(DIM * DIM)), 1e-5f);
  const int mbase = mtile * 16 + (hi ? 8 : 0);
  float av[8];
  #pragma unroll
  for (int j = 0; j < 8; ++j) av[j] = ainv[mbase + j] * wsc;

  const int8_t* aL = smA + l16 * AROW + (hi ? 16 : 0);
  const int colbase = wave * 16 + l16;
  const int8_t* bp = qw + (size_t)colbase * DIM + (hi ? 16 : 0);

  v8i acc[8];
  #pragma unroll
  for (int nt = 0; nt < 8; ++nt) acc[nt] = (v8i){0,0,0,0,0,0,0,0};

  for (int k = 0; k < DIM; k += 64) {
    union { v8i v; int2 p[4]; } A;
    A.p[0] = *(const int2*)(aL + k);
    A.p[1] = *(const int2*)(aL + k + 8);
    A.p[2] = *(const int2*)(aL + k + 32);
    A.p[3] = *(const int2*)(aL + k + 40);
    #pragma unroll
    for (int nt = 0; nt < 8; ++nt) {
      union { v8i v; int4 p[2]; } Bf;
      Bf.p[0] = *(const int4*)(bp + (size_t)nt * 128 * DIM + k);
      Bf.p[1] = *(const int4*)(bp + (size_t)nt * 128 * DIM + k + 32);
      acc[nt] = __builtin_amdgcn_wmma_i32_16x16x64_iu8(true, A.v, true, Bf.v, acc[nt],
                                                       false, false);
    }
  }

  #pragma unroll
  for (int nt = 0; nt < 8; ++nt) {
    const int col = nt * 128 + colbase;
    #pragma unroll
    for (int j = 0; j < 8; ++j) {
      const int m = mbase + j;
      float val = (float)acc[nt][j] * av[j];
      if (mode == 1)      val = sigm(val);
      else if (mode == 2) val = siluf(val) * (1.0f - fprev[(size_t)m * DIM + col]);
      out[(size_t)m * DIM + col] = val;
    }
  }
}

__global__ __launch_bounds__(256) void scan_p1(const float* __restrict__ f,
    const float* __restrict__ in, float* __restrict__ A, float* __restrict__ S) {
  const int tid = blockIdx.x * 256 + threadIdx.x;
  const int d = tid & (DIM - 1);
  const int c = (tid >> 10) & (CH - 1);
  const int b = tid >> 15;
  size_t base = ((size_t)(b * TT + c * CL)) * DIM + d;
  float a = 1.f, s = 0.f;
  for (int t = 0; t < CL; ++t) {
    const float ft = f[base];
    const float it = in[base];
    a *= ft; s = ft * s + it;
    base += DIM;
  }
  A[tid] = a; S[tid] = s;
}
__global__ __launch_bounds__(256) void scan_p2(const float* __restrict__ A,
    const float* __restrict__ S, float* __restrict__ C) {
  const int tid = blockIdx.x * 256 + threadIdx.x;
  const int d = tid & (DIM - 1);
  const int b = tid >> 10;
  float carry = 0.f;
  for (int c = 0; c < CH; ++c) {
    const size_t idx = ((size_t)(b * CH + c)) * DIM + d;
    C[idx] = carry;
    carry = A[idx] * carry + S[idx];
  }
}
__global__ __launch_bounds__(256) void scan_p3(const float* __restrict__ f,
    float* __restrict__ io, const float* __restrict__ C) {
  const int tid = blockIdx.x * 256 + threadIdx.x;
  const int d = tid & (DIM - 1);
  const int c = (tid >> 10) & (CH - 1);
  const int b = tid >> 15;
  size_t base = ((size_t)(b * TT + c * CL)) * DIM + d;
  float h = C[tid];
  for (int t = 0; t < CL; ++t) {
    h = f[base] * h + io[base];
    io[base] = h;
    base += DIM;
  }
}

__global__ __launch_bounds__(256) void gated_out(const float* __restrict__ h,
    const float* __restrict__ g, const float* __restrict__ gnw, float* __restrict__ o) {
  __shared__ float s8[8];
  const int row = blockIdx.x;
  const int t = threadIdx.x;
  const float4 hv = ((const float4*)(h + (size_t)row * DIM))[t];
  float ss = hv.x * hv.x + hv.y * hv.y + hv.z * hv.z + hv.w * hv.w;
  const float tot = blk_sum256(ss, s8);
  const float r = rsqrtf(tot * (1.0f / (float)DIM) + 1e-6f);
  const float4 gv = ((const float4*)(g + (size_t)row * DIM))[t];
  const float4 nv = ((const float4*)gnw)[t];
  float4 ov;
  ov.x = hv.x * r * nv.x * siluf(gv.x);
  ov.y = hv.y * r * nv.y * siluf(gv.y);
  ov.z = hv.z * r * nv.z * siluf(gv.z);
  ov.w = hv.w * r * nv.w * siluf(gv.w);
  ((float4*)(o + (size_t)row * DIM))[t] = ov;
}

__global__ __launch_bounds__(256) void final_rms(const float* __restrict__ x,
    const float* __restrict__ nw, float* __restrict__ out) {
  __shared__ float s8[8];
  const int row = blockIdx.x;
  const int t = threadIdx.x;
  const float4 v = ((const float4*)(x + (size_t)row * DIM))[t];
  float ss = v.x * v.x + v.y * v.y + v.z * v.z + v.w * v.w;
  const float tot = blk_sum256(ss, s8);
  const float r = rsqrtf(tot * (1.0f / (float)DIM) + 1e-6f);
  const float4 nv = ((const float4*)nw)[t];
  float4 ov = {v.x * r * nv.x, v.y * r * nv.y, v.z * r * nv.z, v.w * r * nv.w};
  ((float4*)(out + (size_t)row * DIM))[t] = ov;
}

extern "C" void kernel_launch(void* const* d_in, const int* in_sizes, int n_in,
                              void* d_out, int out_size, void* d_ws, size_t ws_size,
                              hipStream_t stream) {
  const int* tokens = (const int*)d_in[0];
  const float* emb = (const float*)d_in[1];
  const float* Wi = (const float*)d_in[2];
  const float* Wf = (const float*)d_in[3];
  const float* Wg = (const float*)d_in[4];
  const float* Wo = (const float*)d_in[5];
  const float* ni = (const float*)d_in[6];
  const float* nf = (const float*)d_in[7];
  const float* ng = (const float*)d_in[8];
  const float* no = (const float*)d_in[9];
  const float* gnw = (const float*)d_in[10];
  const float* lnw = (const float*)d_in[11];

  char* w = (char*)d_ws;
  float* xbuf = (float*)w; w += (size_t)BT * DIM * 4;
  float* ibuf = (float*)w; w += (size_t)BT * DIM * 4;
  float* fbuf = (float*)w; w += (size_t)BT * DIM * 4;
  float* gbuf = (float*)w; w += (size_t)BT * DIM * 4;
  int8_t* qa  = (int8_t*)w; w += (size_t)BT * DIM;
  int8_t* qw  = (int8_t*)w; w += (size_t)DIM * DIM;
  float* ainv = (float*)w; w += (size_t)BT * 4;
  float* wsum = (float*)w; w += 256;
  float* wpart = (float*)w; w += 256 * 4;
  float* Ab = (float*)w; w += (size_t)BN * CH * DIM * 4;
  float* Sb = (float*)w; w += (size_t)BN * CH * DIM * 4;
  float* Cb = (float*)w; w += (size_t)BN * CH * DIM * 4;

  const int ew = (BT * DIM / 4) / 256;
  embed_kernel<<<ew, 256, 0, stream>>>(tokens, emb, xbuf);

  const int gemm_grid = BT / 16;

  for (int l = 0; l < LN; ++l) {
    const float* WiL = Wi + (size_t)l * DIM * DIM;
    const float* WfL = Wf + (size_t)l * DIM * DIM;
    const float* WgL = Wg + (size_t)l * DIM * DIM;
    const float* WoL = Wo + (size_t)l * DIM * DIM;

    rq_kernel<<<BT, 256, 0, stream>>>(xbuf, nf + l * DIM, qa, ainv);
    wq_part<<<256, 256, 0, stream>>>(WfL, wpart);
    wq_final<<<1, 256, 0, stream>>>(wpart, wsum);
    wq_quant<<<DIM * DIM / 4 / 256, 256, 0, stream>>>(WfL, wsum, qw);
    gemm_kernel<<<gemm_grid, 256, 0, stream>>>(qa, ainv, qw, wsum, fbuf, nullptr, 1);

    rq_kernel<<<BT, 256, 0, stream>>>(xbuf, ni + l * DIM, qa, ainv);
    wq_part<<<256, 256, 0, stream>>>(WiL, wpart);
    wq_final<<<1, 256, 0, stream>>>(wpart, wsum);
    wq_quant<<<DIM * DIM / 4 / 256, 256, 0, stream>>>(WiL, wsum, qw);
    gemm_kernel<<<gemm_grid, 256, 0, stream>>>(qa, ainv, qw, wsum, ibuf, fbuf, 2);

    rq_kernel<<<BT, 256, 0, stream>>>(xbuf, ng + l * DIM, qa, ainv);
    wq_part<<<256, 256, 0, stream>>>(WgL, wpart);
    wq_final<<<1, 256, 0, stream>>>(wpart, wsum);
    wq_quant<<<DIM * DIM / 4 / 256, 256, 0, stream>>>(WgL, wsum, qw);
    gemm_kernel<<<gemm_grid, 256, 0, stream>>>(qa, ainv, qw, wsum, gbuf, nullptr, 0);

    scan_p1<<<BN * CH * DIM / 256, 256, 0, stream>>>(fbuf, ibuf, Ab, Sb);
    scan_p2<<<BN * DIM / 256, 256, 0, stream>>>(Ab, Sb, Cb);
    scan_p3<<<BN * CH * DIM / 256, 256, 0, stream>>>(fbuf, ibuf, Cb);

    gated_out<<<BT, 256, 0, stream>>>(ibuf, gbuf, gnw + l * DIM, fbuf);

    rq_kernel<<<BT, 256, 0, stream>>>(fbuf, no + l * DIM, qa, ainv);
    wq_part<<<256, 256, 0, stream>>>(WoL, wpart);
    wq_final<<<1, 256, 0, stream>>>(wpart, wsum);
    wq_quant<<<DIM * DIM / 4 / 256, 256, 0, stream>>>(WoL, wsum, qw);
    gemm_kernel<<<gemm_grid, 256, 0, stream>>>(qa, ainv, qw, wsum, xbuf, nullptr, 0);
  }

  final_rms<<<BT, 256, 0, stream>>>(xbuf, lnw, (float*)d_out);
}
